// GraphTransformerNet_73418170958214
// MI455X (gfx1250) — compile-verified, matrix operand order checked by reading, not run on hardware
//
#include <hip/hip_runtime.h>
#include <hip/hip_bf16.h>
#include <math.h>

// Problem constants (from reference)
#define G_    256
#define NP_   1000
#define KNN_  8
#define H_    8
#define DH_   10
#define D_    80
#define NL_   10
#define NROW  (G_*NP_)   // 256000

typedef __attribute__((ext_vector_type(16))) _Float16 v16h;
typedef __attribute__((ext_vector_type(8)))  float    v8f;

// GEMM epilogue flags (compile-time template param)
#define F_BIAS 1
#define F_RELU 2
#define F_RES  4

// Async global->LDS staging (gfx1250), guarded so missing builtins fall back
#ifdef __has_builtin
#  if __has_builtin(__builtin_amdgcn_global_load_async_to_lds_b128)
#    define HAS_ASYNC_LDS 1
#  endif
#  if __has_builtin(__builtin_amdgcn_s_wait_asynccnt)
#    define HAS_WAIT_ASYNC 1
#  endif
#endif

// ---------------------------------------------------------------------------
// Weight packer: convert B[K,N] (f32, row-major) into fragment-ordered f16:
//   dst[((t*KSTEPS + ks)*32 + lane)*16 + i]
// matching the CDNA5 wave32 16x16x32 f16 B-fragment layout, zero padded.
// Runs once per launch per weight matrix; GEMM blocks then bulk-copy.
// ---------------------------------------------------------------------------
__global__ void __launch_bounds__(256)
pack_weights_kernel(const float* __restrict__ B, _Float16* __restrict__ dst,
                    int ntiles, int ksteps, int Kreal, int Nreal) {
  int tot = ntiles * ksteps * 512;
  int idx = blockIdx.x * blockDim.x + threadIdx.x;
  if (idx >= tot) return;
  int i   = idx & 15;
  int ln  = (idx >> 4) & 31;
  int f   = idx >> 9;               // fragment id = t*ksteps + ks
  int ks  = f % ksteps;
  int t   = f / ksteps;
  int n   = t * 16 + (ln & 15);
  int klo = ks * 32 + ((ln & 16) ? 8  : 0);
  int khi = ks * 32 + ((ln & 16) ? 24 : 16);
  int k   = (i < 8) ? (klo + i) : (khi + i - 8);
  dst[idx] = (k < Kreal && n < Nreal) ? (_Float16)B[(size_t)k * Nreal + n]
                                      : (_Float16)0.0f;
}

// ---------------------------------------------------------------------------
// A-fragment loader (CDNA5 wave32 16x16x32 f16 A layout, ISA 7.12.2):
//   lanes 0-15  -> M=lane,    K groups [ks*32+0 ..] and [ks*32+16 ..]
//   lanes 16-31 -> M=lane-16, K groups [ks*32+8 ..] and [ks*32+24 ..]
// Each 8-float K group loaded as two float4 when fully in bounds.
// ---------------------------------------------------------------------------
__device__ __forceinline__ v16h load_afrag(const float* __restrict__ Ar,
                                           int Kreal, int ks, int lane) {
  v16h a;
  const int klo = ks * 32 + ((lane & 16) ? 8  : 0);
  const int khi = ks * 32 + ((lane & 16) ? 24 : 16);

  if (klo + 8 <= Kreal) {
    const float4* p = (const float4*)(Ar + klo);
    float4 x = p[0], y = p[1];
    a[0] = (_Float16)x.x; a[1] = (_Float16)x.y; a[2] = (_Float16)x.z; a[3] = (_Float16)x.w;
    a[4] = (_Float16)y.x; a[5] = (_Float16)y.y; a[6] = (_Float16)y.z; a[7] = (_Float16)y.w;
  } else {
#pragma unroll
    for (int i = 0; i < 8; ++i) {
      int k = klo + i;
      a[i] = (k < Kreal) ? (_Float16)Ar[k] : (_Float16)0.0f;
    }
  }
  if (khi + 8 <= Kreal) {
    const float4* p = (const float4*)(Ar + khi);
    float4 x = p[0], y = p[1];
    a[8]  = (_Float16)x.x; a[9]  = (_Float16)x.y; a[10] = (_Float16)x.z; a[11] = (_Float16)x.w;
    a[12] = (_Float16)y.x; a[13] = (_Float16)y.y; a[14] = (_Float16)y.z; a[15] = (_Float16)y.w;
  } else {
#pragma unroll
    for (int i = 0; i < 8; ++i) {
      int k = khi + i;
      a[8 + i] = (k < Kreal) ? (_Float16)Ar[k] : (_Float16)0.0f;
    }
  }
  return a;
}

// ---------------------------------------------------------------------------
// Generic fused GEMM: C[M,N] = op( A[M,K] * Bpacked + bias + R )
// 256 threads = 8 waves; wave w owns rows [ (bid*8+w)*16 , +16 ).
// Grid must cover M exactly (M = grid*128). Bpacked is the fragment-ordered
// f16 weight image; staging is a bulk 16B-chunk copy (async-to-LDS if the
// toolchain exposes the gfx1250 builtin).
// ---------------------------------------------------------------------------
template <int NTILES, int KSTEPS, int FLAGS>
__global__ void __launch_bounds__(256)
wmma_gemm_kernel(const float* __restrict__ A, const _Float16* __restrict__ Bp,
                 const float* __restrict__ bias, const float* __restrict__ R,
                 float* __restrict__ C, int Kreal, int Nreal, int lda) {
  constexpr int TOT = NTILES * KSTEPS * 512;  // halves
  constexpr int CH  = TOT / 8;                // 16-byte chunks
  __shared__ __align__(32) _Float16 Bs[TOT];

  {
    const uint4* src = (const uint4*)Bp;
    uint4* dst = (uint4*)Bs;
#if defined(HAS_ASYNC_LDS)
    typedef __attribute__((__vector_size__(4 * sizeof(int)))) int vint4;
    for (int idx = threadIdx.x; idx < CH; idx += 256) {
      __builtin_amdgcn_global_load_async_to_lds_b128(
          (vint4*)(src + idx), (vint4*)(dst + idx), 0, 0);
    }
#  if defined(HAS_WAIT_ASYNC)
    __builtin_amdgcn_s_wait_asynccnt(0);
#  else
    asm volatile("s_wait_asynccnt 0x0" ::: "memory");
#  endif
#else
    for (int idx = threadIdx.x; idx < CH; idx += 256) dst[idx] = src[idx];
#endif
  }
  __syncthreads();

  const int wave = threadIdx.x >> 5;
  const int lane = threadIdx.x & 31;
  const int mbase = (blockIdx.x * 8 + wave) * 16;
  const float* Ar = A + (size_t)(mbase + (lane & 15)) * lda;

  __builtin_prefetch(Ar, 0, 1);  // global_prefetch_b8

  v8f acc[NTILES];
#pragma unroll
  for (int t = 0; t < NTILES; ++t)
#pragma unroll
    for (int i = 0; i < 8; ++i) acc[t][i] = 0.0f;

#pragma unroll
  for (int ks = 0; ks < KSTEPS; ++ks) {
    v16h a = load_afrag(Ar, Kreal, ks, lane);
#pragma unroll
    for (int t = 0; t < NTILES; ++t) {
      const v16h* bp = reinterpret_cast<const v16h*>(
          Bs + ((t * KSTEPS + ks) * 32 + lane) * 16);
      v16h b = *bp;                          // 2x ds_load_b128
      acc[t] = __builtin_amdgcn_wmma_f32_16x16x32_f16(
          /*neg_a=*/false, a, /*neg_b=*/false, b,
          /*c_mod=*/(short)0, acc[t], /*reuse_a=*/false, /*reuse_b=*/false);
    }
  }

  // Epilogue. C/D layout: VGPR r -> M = r (+8 for lanes>=16); N = lane&15.
  // Lanes 0-15 of a wave cover 16 consecutive columns of one row -> coalesced.
  const int rowoff = (lane & 16) ? 8 : 0;
  const int ncol0  = lane & 15;
#pragma unroll
  for (int t = 0; t < NTILES; ++t) {
    int col = t * 16 + ncol0;
    if (col < Nreal) {
      float bv = (FLAGS & F_BIAS) ? bias[col] : 0.0f;
      int off0 = (mbase + rowoff) * Nreal + col;
#pragma unroll
      for (int r = 0; r < 8; ++r) {
        int off = off0 + r * Nreal;
        float v = acc[t][r] + bv;
        if (FLAGS & F_RES)  v += R[off];
        if (FLAGS & F_RELU) v = fmaxf(v, 0.0f);
        C[off] = v;
      }
    }
  }
}

// ---------------------------------------------------------------------------
// kNN: per graph, top-8 smallest squared distances (stable, includes self).
// One block per graph; coords staged in LDS.
// ---------------------------------------------------------------------------
__global__ void __launch_bounds__(256)
knn_kernel(const float* __restrict__ h_feat, int* __restrict__ nbr) {
  __shared__ float sx[NP_], sy[NP_], sz[NP_];
  const int g = blockIdx.x;
  const float* base = h_feat + (size_t)g * NP_ * 9;
  for (int n = threadIdx.x; n < NP_; n += blockDim.x) {
    sx[n] = base[n * 9 + 0];
    sy[n] = base[n * 9 + 1];
    sz[n] = base[n * 9 + 2];
  }
  __syncthreads();

  for (int n = threadIdx.x; n < NP_; n += blockDim.x) {
    float px = sx[n], py = sy[n], pz = sz[n];
    float bd[KNN_];
    int   bi[KNN_];
#pragma unroll
    for (int p = 0; p < KNN_; ++p) { bd[p] = 3.4e38f; bi[p] = 0; }
    for (int j = 0; j < NP_; ++j) {
      float dx = px - sx[j], dy = py - sy[j], dz = pz - sz[j];
      float d2 = dx * dx + dy * dy + dz * dz;
      // unrolled stable insertion (strict < keeps earlier index on ties,
      // matching jax.lax.top_k stability)
      float cd = d2; int ci = j;
#pragma unroll
      for (int p = 0; p < KNN_; ++p) {
        bool ins = cd < bd[p];
        float td = ins ? bd[p] : cd;
        int   ti = ins ? bi[p] : ci;
        bd[p] = ins ? cd : bd[p];
        bi[p] = ins ? ci : bi[p];
        cd = td; ci = ti;
      }
    }
    int* o = nbr + ((size_t)g * NP_ + n) * KNN_;
#pragma unroll
    for (int p = 0; p < KNN_; ++p) o[p] = bi[p];
  }
}

// ---------------------------------------------------------------------------
// BatchNorm statistics: sums[0..8]=sum, sums[9..17]=sum of squares
// ---------------------------------------------------------------------------
__global__ void zero_kernel(float* __restrict__ p, int n) {
  int i = blockIdx.x * blockDim.x + threadIdx.x;
  if (i < n) p[i] = 0.0f;
}

__global__ void __launch_bounds__(256)
bn_sums_kernel(const float* __restrict__ x, float* __restrict__ sums) {
  __shared__ float red[256];
  float acc[18];
#pragma unroll
  for (int f = 0; f < 18; ++f) acc[f] = 0.0f;
  int total = gridDim.x * blockDim.x;
  for (int row = blockIdx.x * blockDim.x + threadIdx.x; row < NROW; row += total) {
    const float* r = x + (size_t)row * 9;
#pragma unroll
    for (int f = 0; f < 9; ++f) {
      float v = r[f];
      acc[f]     += v;
      acc[9 + f] += v * v;
    }
  }
  for (int f = 0; f < 18; ++f) {
    red[threadIdx.x] = acc[f];
    __syncthreads();
    for (int s = 128; s > 0; s >>= 1) {
      if (threadIdx.x < s) red[threadIdx.x] += red[threadIdx.x + s];
      __syncthreads();
    }
    if (threadIdx.x == 0) atomicAdd(&sums[f], red[0]);
    __syncthreads();
  }
}

// Normalize with zero-padded row stride 16:
// xn[row*16 + f] = (x - mu) * rsqrt(var + 1e-5) * gamma + beta   (f < 9)
// xn[row*16 + f] = 0                                             (9 <= f < 16)
__global__ void bn_norm_kernel(const float* __restrict__ x,
                               const float* __restrict__ sums,
                               const float* __restrict__ gamma,
                               const float* __restrict__ beta,
                               float* __restrict__ xn) {
  int idx = blockIdx.x * blockDim.x + threadIdx.x;
  if (idx >= NROW * 16) return;
  int row = idx >> 4;
  int f   = idx & 15;
  float v = 0.0f;
  if (f < 9) {
    const float invN = 1.0f / (float)NROW;
    float mu  = sums[f] * invN;
    float var = sums[9 + f] * invN - mu * mu;
    v = (x[(size_t)row * 9 + f] - mu) * rsqrtf(var + 1e-5f) * gamma[f] + beta[f];
  }
  xn[idx] = v;
}

// ---------------------------------------------------------------------------
// Neighbor attention: one thread per (g, n, head)
// ---------------------------------------------------------------------------
__global__ void __launch_bounds__(256)
attn_kernel(const float* __restrict__ Q, const float* __restrict__ Kb,
            const float* __restrict__ V, const int* __restrict__ nbr,
            float* __restrict__ attn) {
  int idx = blockIdx.x * blockDim.x + threadIdx.x;
  if (idx >= NROW * H_) return;
  int h   = idx % H_;
  int row = idx / H_;          // g*NP + n
  int g   = row / NP_;
  const float inv_sqrt_d = 0.31622776601683794f;  // 1/sqrt(10)

  float q[DH_];
  const float* qr = Q + (size_t)row * D_ + h * DH_;
#pragma unroll
  for (int d = 0; d < DH_; ++d) q[d] = qr[d];

  float wv[DH_];
#pragma unroll
  for (int d = 0; d < DH_; ++d) wv[d] = 0.0f;
  float ssum = 0.0f;

  const int* nb = nbr + (size_t)row * KNN_;
#pragma unroll
  for (int k = 0; k < KNN_; ++k) {
    int j  = nb[k];
    size_t jr = ((size_t)g * NP_ + j) * D_ + h * DH_;
    float s = 0.0f;
#pragma unroll
    for (int d = 0; d < DH_; ++d) s += Kb[jr + d] * q[d];
    s *= inv_sqrt_d;
    s = fminf(fmaxf(s, -5.0f), 5.0f);
    s = expf(s);
    ssum += s;
#pragma unroll
    for (int d = 0; d < DH_; ++d) wv[d] += s * V[jr + d];
  }
  float inv = 1.0f / (ssum + 1e-6f);
  float* o = attn + (size_t)row * D_ + h * DH_;
#pragma unroll
  for (int d = 0; d < DH_; ++d) o[d] = wv[d] * inv;
}

// ---------------------------------------------------------------------------
// Head: out[:,0:3] = y @ W_clust ; out[:,3] = y @ W_beta + b_beta
// ---------------------------------------------------------------------------
__global__ void __launch_bounds__(256)
head_kernel(const float* __restrict__ Y, const float* __restrict__ Wc,
            const float* __restrict__ Wb, const float* __restrict__ bb,
            float* __restrict__ out, int M) {
  int row = blockIdx.x * blockDim.x + threadIdx.x;
  if (row >= M) return;
  const float* y = Y + (size_t)row * 64;
  float c0 = 0.f, c1 = 0.f, c2 = 0.f, bsum = 0.f;
#pragma unroll
  for (int i = 0; i < 64; ++i) {
    float v = y[i];
    c0 += v * Wc[i * 3 + 0];
    c1 += v * Wc[i * 3 + 1];
    c2 += v * Wc[i * 3 + 2];
    bsum += v * Wb[i];
  }
  float* o = out + (size_t)row * 4;
  o[0] = c0; o[1] = c1; o[2] = c2; o[3] = bsum + bb[0];
}

// ---------------------------------------------------------------------------
// Host-side orchestration
// ---------------------------------------------------------------------------
extern "C" void kernel_launch(void* const* d_in, const int* in_sizes, int n_in,
                              void* d_out, int out_size, void* d_ws, size_t ws_size,
                              hipStream_t stream) {
  (void)in_sizes; (void)n_in; (void)out_size; (void)ws_size;

  const float* h_feat   = (const float*)d_in[0];
  const float* bn_gamma = (const float*)d_in[1];
  const float* bn_beta  = (const float*)d_in[2];
  const float* W_emb    = (const float*)d_in[3];
  const float* b_emb    = (const float*)d_in[4];
  const float* Wq       = (const float*)d_in[5];
  const float* Wk       = (const float*)d_in[6];
  const float* Wv       = (const float*)d_in[7];
  const float* Wo       = (const float*)d_in[8];
  const float* bo       = (const float*)d_in[9];
  const float* W1       = (const float*)d_in[10];
  const float* b1       = (const float*)d_in[11];
  const float* W2       = (const float*)d_in[12];
  const float* b2       = (const float*)d_in[13];
  const float* Wr0      = (const float*)d_in[14];
  const float* br0      = (const float*)d_in[15];
  const float* Wr1      = (const float*)d_in[16];
  const float* br1      = (const float*)d_in[17];
  const float* Wr2      = (const float*)d_in[18];
  const float* br2      = (const float*)d_in[19];
  const float* W_clust  = (const float*)d_in[20];
  const float* W_beta   = (const float*)d_in[21];
  const float* b_beta   = (const float*)d_in[22];
  float* out = (float*)d_out;

  // Workspace carve-out (deterministic layout each call)
  char* ws = (char*)d_ws;
  size_t off = 0;
  auto carve = [&](size_t bytes) -> void* {
    void* p = ws + off;
    off = (off + bytes + 255) & ~(size_t)255;
    return p;
  };
  float* sums = (float*)carve(18 * sizeof(float));
  int*   nbr  = (int*)  carve((size_t)NROW * KNN_ * sizeof(int));
  float* hA   = (float*)carve((size_t)NROW * D_ * sizeof(float));      // h
  float* hB   = (float*)carve((size_t)NROW * D_ * sizeof(float));      // h2
  float* Qb   = (float*)carve((size_t)NROW * D_ * sizeof(float));
  float* Kbuf = (float*)carve((size_t)NROW * D_ * sizeof(float));
  float* Vb   = (float*)carve((size_t)NROW * D_ * sizeof(float));
  float* Sb   = (float*)carve((size_t)NROW * 2 * D_ * sizeof(float));  // xn/attn/ffn-hidden

  // Packed f16 weight images (fragment order), produced once per launch.
  auto pack = [&](const float* B, int nt, int ks, int Kreal, int Nreal) -> const _Float16* {
    size_t tot = (size_t)nt * ks * 512;
    _Float16* dst = (_Float16*)carve(tot * sizeof(_Float16));
    pack_weights_kernel<<<(int)((tot + 255) / 256), 256, 0, stream>>>(
        B, dst, nt, ks, Kreal, Nreal);
    return dst;
  };

  const _Float16* pemb = pack(W_emb, 5, 1, 9, D_);
  const _Float16 *pq[NL_], *pk[NL_], *pv[NL_], *po[NL_], *p1[NL_], *p2[NL_];
  for (int l = 0; l < NL_; ++l) {
    pq[l] = pack(Wq + (size_t)l * D_ * D_,     5, 3, D_,     D_);
    pk[l] = pack(Wk + (size_t)l * D_ * D_,     5, 3, D_,     D_);
    pv[l] = pack(Wv + (size_t)l * D_ * D_,     5, 3, D_,     D_);
    po[l] = pack(Wo + (size_t)l * D_ * D_,     5, 3, D_,     D_);
    p1[l] = pack(W1 + (size_t)l * D_ * 2*D_,  10, 3, D_,     2*D_);
    p2[l] = pack(W2 + (size_t)l * 2*D_ * D_,   5, 5, 2*D_,   D_);
  }
  const _Float16* pr0 = pack(Wr0, 3, 3, D_, 40);
  const _Float16* pr1 = pack(Wr1, 2, 2, 40, 20);
  const _Float16* pr2 = pack(Wr2, 4, 1, 20, 64);

  const int GEMM_GRID = NROW / 128;  // exact: 8 waves * 16 rows per block

  // 1) Preprocess: BN stats, kNN, normalize (padded to stride 16)
  zero_kernel<<<1, 32, 0, stream>>>(sums, 18);
  bn_sums_kernel<<<512, 256, 0, stream>>>(h_feat, sums);
  knn_kernel<<<G_, 256, 0, stream>>>(h_feat, nbr);
  bn_norm_kernel<<<(NROW * 16 + 255) / 256, 256, 0, stream>>>(
      h_feat, sums, bn_gamma, bn_beta, Sb);

  // 2) Embedding: hA = xn(256000x9, lda=16) @ W_emb(9x80) + b_emb
  wmma_gemm_kernel<5, 1, F_BIAS><<<GEMM_GRID, 256, 0, stream>>>(
      Sb, pemb, b_emb, nullptr, hA, 9, D_, 16);

  // 3) Transformer layers
  for (int l = 0; l < NL_; ++l) {
    const float* bol = bo + (size_t)l * D_;
    const float* b1l = b1 + (size_t)l * 2 * D_;
    const float* b2l = b2 + (size_t)l * D_;

    // q,k,v = h @ W{q,k,v}
    wmma_gemm_kernel<5, 3, 0><<<GEMM_GRID, 256, 0, stream>>>(
        hA, pq[l], nullptr, nullptr, Qb, D_, D_, D_);
    wmma_gemm_kernel<5, 3, 0><<<GEMM_GRID, 256, 0, stream>>>(
        hA, pk[l], nullptr, nullptr, Kbuf, D_, D_, D_);
    wmma_gemm_kernel<5, 3, 0><<<GEMM_GRID, 256, 0, stream>>>(
        hA, pv[l], nullptr, nullptr, Vb, D_, D_, D_);

    // neighbor attention -> Sb (stride D_)
    attn_kernel<<<(NROW * H_ + 255) / 256, 256, 0, stream>>>(Qb, Kbuf, Vb, nbr, Sb);

    // h2 = h + attn @ Wo + bo
    wmma_gemm_kernel<5, 3, F_BIAS | F_RES><<<GEMM_GRID, 256, 0, stream>>>(
        Sb, po[l], bol, hA, hB, D_, D_, D_);

    // t = relu(h2 @ W1 + b1)   (256000 x 160)
    wmma_gemm_kernel<10, 3, F_BIAS | F_RELU><<<GEMM_GRID, 256, 0, stream>>>(
        hB, p1[l], b1l, nullptr, Sb, D_, 2 * D_, D_);

    // h = h2 + t @ W2 + b2
    wmma_gemm_kernel<5, 5, F_BIAS | F_RES><<<GEMM_GRID, 256, 0, stream>>>(
        Sb, p2[l], b2l, hB, hA, 2 * D_, D_, 2 * D_);
  }

  // 4) Readout MLP
  wmma_gemm_kernel<3, 3, F_BIAS | F_RELU><<<GEMM_GRID, 256, 0, stream>>>(
      hA, pr0, br0, nullptr, Qb, D_, 40, D_);
  wmma_gemm_kernel<2, 2, F_BIAS | F_RELU><<<GEMM_GRID, 256, 0, stream>>>(
      Qb, pr1, br1, nullptr, Kbuf, 40, 20, 40);
  wmma_gemm_kernel<4, 1, F_BIAS><<<GEMM_GRID, 256, 0, stream>>>(
      Kbuf, pr2, br2, nullptr, Vb, 20, 64, 20);

  // 5) Heads -> out (NROW x 4)
  head_kernel<<<(NROW + 255) / 256, 256, 0, stream>>>(
      Vb, W_clust, W_beta, b_beta, out, NROW);
}
